// gru_model_446676598973
// MI455X (gfx1250) — matrix operand order repaired and checked by reading, not compiled, not run on hardware
//
#include <hip/hip_runtime.h>

// ---------------- model constants ----------------
#define CB   128      // batch
#define CT   288      // time
#define CU   256      // units
#define CH   4        // heads
#define CHD  64       // head dim
#define CMT  (CB*CT)  // 36864 rows
#define C3U  768

typedef __attribute__((ext_vector_type(16))) __bf16 v16bf;
typedef __attribute__((ext_vector_type(8)))  float  v8f;

union Frag128x2 { v16bf v; uint4 u[2]; };

// A-matrix 16x32 bf16 fragment (contiguous K, matching load_b16):
// lanes 0-15 row M=lane, K=0..15 (kbase=0)
// lanes 16-31 same rows, K=16..31 (kbase=8).
__device__ __forceinline__ v16bf load_a16(const __bf16* rowp_k0, int kbase) {
  Frag128x2 f;
  f.u[0] = *reinterpret_cast<const uint4*>(rowp_k0 + 2 * kbase);
  f.u[1] = *reinterpret_cast<const uint4*>(rowp_k0 + 2 * kbase + 8);
  return f.v;
}
// B-matrix 32x16 bf16 fragment: lane = column n, lanes 0-15 hold K=0..15,
// lanes 16-31 hold K=16..31 (contiguous along K in [N,K]-transposed storage).
__device__ __forceinline__ v16bf load_b16(const __bf16* col_k0_khalf) {
  Frag128x2 f;
  const uint4* p = reinterpret_cast<const uint4*>(col_k0_khalf);
  f.u[0] = p[0];
  f.u[1] = p[1];
  return f.v;
}

__device__ __forceinline__ v8f wmma_bf16(v16bf a, v16bf b, v8f c) {
  return __builtin_amdgcn_wmma_f32_16x16x32_bf16(false, a, false, b, (short)0, c, false, false);
}

// ---------------- CDNA5 async global->LDS copy (ASYNCcnt path) ----------------
// Per-lane: LDS[dsaddr..+15] = MEM[vaddr..+15]; dsaddr is low 32 bits of the
// generic LDS pointer (ISA: flat LDS addr truncated to addr[31:0]).
__device__ __forceinline__ void async_b128(const void* gsrc, void* ldst) {
  asm volatile("global_load_async_to_lds_b128 %0, %1, off"
               :: "v"((unsigned)(uintptr_t)ldst), "v"(gsrc)
               : "memory");
}
#define ASYNC_WAIT(n) asm volatile("s_wait_asynccnt " #n ::: "memory")

// ---------------- block reduction (wave32) ----------------
__device__ __forceinline__ float block_sum(float v, float* sbuf) {
#pragma unroll
  for (int m = 16; m >= 1; m >>= 1) v += __shfl_xor(v, m, 32);
  int lane = threadIdx.x & 31, wid = threadIdx.x >> 5;
  int nw = (blockDim.x + 31) >> 5;
  if (lane == 0) sbuf[wid] = v;
  __syncthreads();
  float r = (threadIdx.x < (unsigned)nw) ? sbuf[threadIdx.x] : 0.f;
#pragma unroll
  for (int m = 16; m >= 1; m >>= 1) r += __shfl_xor(r, m, 32);
  if (threadIdx.x == 0) sbuf[0] = r;
  __syncthreads();
  r = sbuf[0];
  __syncthreads();
  return r;
}

__device__ __forceinline__ float sigmoidf_(float x) { return 1.f / (1.f + __expf(-x)); }

// ---------------- weight prep ----------------
// src [K,N] f32 -> dst [N,K] bf16
__global__ void k_transpose_bf16(const float* __restrict__ src, __bf16* __restrict__ dst,
                                 int K, int N) {
  int idx = blockIdx.x * 256 + threadIdx.x;
  if (idx >= K * N) return;
  int n = idx / K, k = idx % K;
  dst[idx] = (__bf16)src[(size_t)k * N + n];
}

__global__ void k_pack_qkv_bias(const float* __restrict__ bq, const float* __restrict__ bk,
                                const float* __restrict__ bv, float* __restrict__ dst) {
  int n = blockIdx.x * 256 + threadIdx.x;  // 0..767
  int i = blockIdx.y;                       // block index
  const float* s3[3] = {bq, bk, bv};
  dst[i * C3U + n] = s3[n >> 8][i * CU + (n & 255)];
}

// ---------------- input projection + LN0 ----------------
__global__ __launch_bounds__(256)
void k_inproj_ln(const float* __restrict__ cgm, const float* __restrict__ W,
                 const float* __restrict__ bvec, const float* __restrict__ s,
                 const float* __restrict__ bb, float* __restrict__ xf,
                 __bf16* __restrict__ xbf) {
  __shared__ float red[32];
  int row = blockIdx.x;        // b*T + t
  int u = threadIdx.x;         // 0..255
  const float* cg = cgm + (size_t)row * 16;
  float acc = bvec[u];
#pragma unroll
  for (int k = 0; k < 16; k++) acc += cg[k] * W[k * CU + u];
  float mean = block_sum(acc, red) * (1.f / CU);
  float d = acc - mean;
  float var = block_sum(d * d, red) * (1.f / CU);
  float y = d * rsqrtf(var + 1e-6f) * s[u] + bb[u];
  size_t idx = (size_t)row * CU + u;
  xf[idx] = y;
  xbf[idx] = (__bf16)y;
}

// ---------------- residual add + LN (writes f32 + bf16) ----------------
__global__ __launch_bounds__(256)
void k_add_ln(const float* __restrict__ a, const float* __restrict__ skip,
              const float* __restrict__ s, const float* __restrict__ bb,
              float* __restrict__ xf, __bf16* __restrict__ xbf) {
  __shared__ float red[32];
  int row = blockIdx.x;
  int u = threadIdx.x;
  size_t idx = (size_t)row * CU + u;
  float v = a[idx] + skip[idx];
  float mean = block_sum(v, red) * (1.f / CU);
  float d = v - mean;
  float var = block_sum(d * d, red) * (1.f / CU);
  float y = d * rsqrtf(var + 1e-6f) * s[u] + bb[u];
  xf[idx] = y;
  xbf[idx] = (__bf16)y;
}

// ---------------- WMMA GEMM: C[M,N] = A[M,K](bf16) @ Wt[N,K](bf16) + bias ----------------
// B panel (64 cols x K) is async-staged into LDS once per block (shared by all
// 8 waves -> 8x less global B traffic). K-loop is unrolled by 2 with ping-pong
// fragment sets: set A is loop-carried but only written by loads (no register
// rotation movs), set B is iteration-local. Loads for each WMMA batch are in
// flight half a step early.
// mode 0: write f32 to Cf[M,N]
// mode 1: QKV scatter to q[B,H,T,64], k[B,H,T,64], vT[B,H,64,T] as bf16
__global__ __launch_bounds__(256)
void k_gemm_bf16(const __bf16* __restrict__ A, const __bf16* __restrict__ Wt,
                 const float* __restrict__ bias, float* __restrict__ Cf,
                 __bf16* __restrict__ qb, __bf16* __restrict__ kbp,
                 __bf16* __restrict__ vtb, int M, int N, int K, int mode) {
  __shared__ __align__(16) __bf16 Bp[64 * CU];   // 32 KB, K must be 256
  int tid = threadIdx.x;
  int lane = tid & 31, wv = tid >> 5;
  int m0 = blockIdx.y * 128 + wv * 16;
  int n0 = blockIdx.x * 64;

  // async stage of the contiguous B panel Wt[n0..n0+63][0..K)
  {
    const char* gpanel = (const char*)(Wt + (size_t)n0 * K);
    char* lpanel = (char*)Bp;
    const int bytes = 64 * CU * 2;
    for (int o = tid * 16; o < bytes; o += 256 * 16)
      async_b128(gpanel + o, lpanel + o);
  }
  ASYNC_WAIT(0);
  __syncthreads();

  int kbase = (lane < 16) ? 0 : 8;
  int khalf = (lane < 16) ? 0 : 16;
  int rowA = m0 + (lane & 15);
  const __bf16* arow = A + (size_t)rowA * K;
  const __bf16* bcol = Bp + (size_t)(lane & 15) * CU + khalf;
  v8f acc[4] = {};

  // prologue: fragments for K-step 0
  v16bf afA = load_a16(arow, kbase);
  v16bf bfA[4];
#pragma unroll
  for (int nt = 0; nt < 4; nt++) bfA[nt] = load_b16(bcol + (size_t)nt * 16 * CU);

  for (int k0 = 0; k0 < K; k0 += 64) {
    // fragments for k0+32 (set B, iteration-local)
    v16bf afB = load_a16(arow + k0 + 32, kbase);
    v16bf bfB[4];
#pragma unroll
    for (int nt = 0; nt < 4; nt++)
      bfB[nt] = load_b16(bcol + (size_t)nt * 16 * CU + k0 + 32);
    // compute k0 with set A
#pragma unroll
    for (int nt = 0; nt < 4; nt++) acc[nt] = wmma_bf16(afA, bfA[nt], acc[nt]);
    // refill set A with k0+64 fragments
    if (k0 + 64 < K) {
      __builtin_prefetch(arow + k0 + 128, 0, 1);   // global_prefetch_b8
      afA = load_a16(arow + k0 + 64, kbase);
#pragma unroll
      for (int nt = 0; nt < 4; nt++)
        bfA[nt] = load_b16(bcol + (size_t)nt * 16 * CU + k0 + 64);
    }
    // compute k0+32 with set B
#pragma unroll
    for (int nt = 0; nt < 4; nt++) acc[nt] = wmma_bf16(afB, bfB[nt], acc[nt]);
  }

  int rl = (lane >> 4) << 3;     // +0 rows 0-7 (lanes 0-15), +8 rows 8-15 (lanes 16-31)
  int cl = lane & 15;
#pragma unroll
  for (int nt = 0; nt < 4; nt++) {
#pragma unroll
    for (int g = 0; g < 8; g++) {
      int m = m0 + g + rl;
      int n = n0 + nt * 16 + cl;
      float v = acc[nt][g] + bias[n];
      if (mode == 0) {
        Cf[(size_t)m * N + n] = v;
      } else {
        int b = m / CT, t = m % CT;
        int part = n >> 8, u = n & 255;
        int h = u >> 6, d2 = u & 63;
        size_t bh = (size_t)b * CH + h;
        if (part == 0)       qb[(bh * CT + t) * CHD + d2] = (__bf16)v;
        else if (part == 1)  kbp[(bh * CT + t) * CHD + d2] = (__bf16)v;
        else                 vtb[(bh * CHD + d2) * CT + t] = (__bf16)v;
      }
    }
  }
}

// ---------------- GRU sequential scan ----------------
// grid = 8 blocks (16 batch rows each), 512 threads = 16 waves.
// per step: hg = h @ Wh via WMMA (h bf16 in LDS), then gate math, h update.
// The xg slab for step t+1 (16 rows x 768 f32 = 48 KB) is async-prefetched
// into a double-buffered LDS region while step t's WMMA runs. The K-loop uses
// the same ping-pong unroll-by-2 as the GEMM.
__global__ __launch_bounds__(512)
void k_gru_scan(const float* __restrict__ xg, const __bf16* __restrict__ WhT,
                const float* __restrict__ bhn, float* __restrict__ gout) {
  __shared__ __align__(16) float  hS[16][CU];        // 16 KB
  __shared__ __align__(16) __bf16 hbS[16][CU + 8];   // 8.25 KB (pitch 264, 16B mult)
  __shared__ __align__(16) __bf16 hgS[16][C3U + 8];  // 24.25 KB
  __shared__ __align__(16) float  xgS[2][16][C3U];   // 96 KB double buffer
  int tid = threadIdx.x, lane = tid & 31, wv = tid >> 5;
  int b0 = blockIdx.x * 16;
  for (int i = tid; i < 16 * CU; i += 512) {
    hS[i >> 8][i & 255] = 0.f;
    hbS[i >> 8][i & 255] = (__bf16)0.f;
  }
  // stage slab for t = 0 : 49152 bytes / (512 thr * 16B) = 6 issues per thread
#pragma unroll
  for (int it = 0; it < 6; it++) {
    int o = tid * 16 + it * 8192;
    int r = o / 3072, col = o - r * 3072;          // 3072 bytes per row
    const char* ga = (const char*)(xg + ((size_t)(b0 + r) * CT + 0) * C3U) + col;
    async_b128(ga, (char*)&xgS[0][0][0] + o);
  }
  ASYNC_WAIT(0);
  __syncthreads();

  int kbase = (lane < 16) ? 0 : 8;
  int khalf = (lane < 16) ? 0 : 16;
  int rowA = lane & 15;
  int rl = (lane >> 4) << 3, cl = lane & 15;
  int grow = tid >> 5;          // gate row 0..15
  int u0 = (tid & 31) * 8;      // 8 units per thread
  const __bf16* wcol = WhT + (size_t)(wv * 48 + (lane & 15)) * CU + khalf;
  for (int t = 0; t < CT; t++) {
    // kick off async prefetch of next step's xg slab
    if (t + 1 < CT) {
#pragma unroll
      for (int it = 0; it < 6; it++) {
        int o = tid * 16 + it * 8192;
        int r = o / 3072, col = o - r * 3072;
        const char* ga = (const char*)(xg + ((size_t)(b0 + r) * CT + (t + 1)) * C3U) + col;
        async_b128(ga, (char*)&xgS[(t + 1) & 1][0][0] + o);
      }
    }
    // phase 1: hg = h @ Wh  (each wave: 16 rows x 48 cols), ping-pong fragments
    v8f acc[3] = {};
    v16bf afA = load_a16(hbS[rowA], kbase);
    v16bf bwA[3];
#pragma unroll
    for (int nt = 0; nt < 3; nt++) bwA[nt] = load_b16(wcol + (size_t)nt * 16 * CU);
    for (int k0 = 0; k0 < CU; k0 += 64) {
      v16bf afB = load_a16(hbS[rowA] + k0 + 32, kbase);
      v16bf bwB[3];
#pragma unroll
      for (int nt = 0; nt < 3; nt++)
        bwB[nt] = load_b16(wcol + (size_t)nt * 16 * CU + k0 + 32);
#pragma unroll
      for (int nt = 0; nt < 3; nt++) acc[nt] = wmma_bf16(afA, bwA[nt], acc[nt]);
      if (k0 + 64 < CU) {
        afA = load_a16(hbS[rowA] + k0 + 64, kbase);
#pragma unroll
        for (int nt = 0; nt < 3; nt++)
          bwA[nt] = load_b16(wcol + (size_t)nt * 16 * CU + k0 + 64);
      }
#pragma unroll
      for (int nt = 0; nt < 3; nt++) acc[nt] = wmma_bf16(afB, bwB[nt], acc[nt]);
    }
#pragma unroll
    for (int nt = 0; nt < 3; nt++)
#pragma unroll
      for (int g = 0; g < 8; g++)
        hgS[g + rl][wv * 48 + nt * 16 + cl] = (__bf16)acc[nt][g];
    // slab t was issued one iteration ago; async loads complete in order, so
    // allowing the 6 just-issued ops to remain outstanding is sufficient.
    if (t + 1 < CT) { ASYNC_WAIT(6); } else { ASYNC_WAIT(0); }
    __syncthreads();
    // phase 2: gates for 16 rows x 256 units
    const float* xgrow = &xgS[t & 1][grow][0];
    float* gorow = gout + ((size_t)(b0 + grow) * CT + t) * CU;
#pragma unroll
    for (int j = 0; j < 8; j++) {
      int u = u0 + j;
      float r = sigmoidf_(xgrow[u] + (float)hgS[grow][u]);
      float z = sigmoidf_(xgrow[CU + u] + (float)hgS[grow][CU + u]);
      float nn = tanhf(xgrow[2 * CU + u] + r * ((float)hgS[grow][2 * CU + u] + bhn[u]));
      float hprev = hS[grow][u];
      float hnew = (1.f - z) * nn + z * hprev;
      hS[grow][u] = hnew;
      hbS[grow][u] = (__bf16)hnew;
      gorow[u] = hnew;
    }
    __syncthreads();
  }
}

// ---------------- attention ----------------
// grid = (B*H, 3); block = 192 threads = 6 waves; wave = one 16-row q tile.
// K and V^T panels (36 KB each) are async-staged into LDS, shared by all waves.
__global__ __launch_bounds__(192)
void k_attn(const __bf16* __restrict__ qb, const __bf16* __restrict__ kbp,
            const __bf16* __restrict__ vtb, __bf16* __restrict__ obf) {
  __shared__ __align__(16) __bf16 pS[6][16][CT];  // 54 KB
  __shared__ __align__(16) __bf16 kS[CT * CHD];   // 36 KB
  __shared__ __align__(16) __bf16 vS[CHD * CT];   // 36 KB
  int tid = threadIdx.x;
  int lane = tid & 31, wv = tid >> 5;
  int bh = blockIdx.x;
  int qt = blockIdx.y * 6 + wv;     // 0..17
  int b = bh >> 2, h = bh & 3;
  const __bf16* qbase = qb + (size_t)bh * CT * CHD;
  {
    const char* gk = (const char*)(kbp + (size_t)bh * CT * CHD);
    const char* gv = (const char*)(vtb + (size_t)bh * CHD * CT);
    const int bytes = CT * CHD * 2;  // 36864
    for (int o = tid * 16; o < bytes; o += 192 * 16) {
      async_b128(gk + o, (char*)kS + o);
      async_b128(gv + o, (char*)vS + o);
    }
  }
  ASYNC_WAIT(0);
  __syncthreads();

  int kb8 = (lane < 16) ? 0 : 8;
  int kh16 = (lane < 16) ? 0 : 16;
  int tq = qt * 16 + (lane & 15);
  v8f acc[18];
#pragma unroll
  for (int i = 0; i < 18; i++) acc[i] = v8f{};
  // scores = q @ k^T  (K dim = 64 head dims)
#pragma unroll
  for (int k0 = 0; k0 < CHD; k0 += 32) {
    v16bf af = load_a16(qbase + (size_t)tq * CHD + k0, kb8);
#pragma unroll
    for (int kt = 0; kt < 18; kt++) {
      int tk = kt * 16 + (lane & 15);
      v16bf bfv = load_b16(kS + (size_t)tk * CHD + k0 + kh16);
      acc[kt] = wmma_bf16(af, bfv, acc[kt]);
    }
  }
  // softmax over the 288 columns of each row (row fixed within a 16-lane half)
#pragma unroll
  for (int g = 0; g < 8; g++) {
    float mx = -1e30f;
#pragma unroll
    for (int kt = 0; kt < 18; kt++) {
      float v = acc[kt][g] * 0.125f;   // 1/sqrt(64)
      acc[kt][g] = v;
      mx = fmaxf(mx, v);
    }
#pragma unroll
    for (int msk = 8; msk >= 1; msk >>= 1) mx = fmaxf(mx, __shfl_xor(mx, msk, 16));
    float sum = 0.f;
#pragma unroll
    for (int kt = 0; kt < 18; kt++) {
      float e = __expf(acc[kt][g] - mx);
      acc[kt][g] = e;
      sum += e;
    }
#pragma unroll
    for (int msk = 8; msk >= 1; msk >>= 1) sum += __shfl_xor(sum, msk, 16);
    float inv = 1.f / sum;
#pragma unroll
    for (int kt = 0; kt < 18; kt++) acc[kt][g] *= inv;
  }
  // stage p (bf16) to LDS so it can be re-read in A-fragment layout
  int rl = (lane >> 4) << 3, cl = lane & 15;
#pragma unroll
  for (int g = 0; g < 8; g++)
#pragma unroll
    for (int kt = 0; kt < 18; kt++)
      pS[wv][g + rl][kt * 16 + cl] = (__bf16)acc[kt][g];
  __syncthreads();
  // o = p @ v   (K dim = 288 time steps)
  v8f oa[4] = {};
  int rowp = lane & 15;
  const __bf16* vcol = vS + (size_t)(lane & 15) * CT + kh16;
  for (int k0 = 0; k0 < CT; k0 += 32) {
    v16bf af = load_a16(pS[wv][rowp] + k0, kb8);
#pragma unroll
    for (int nt = 0; nt < 4; nt++) {
      v16bf bfv = load_b16(vcol + (size_t)nt * 16 * CT + k0);
      oa[nt] = wmma_bf16(af, bfv, oa[nt]);
    }
  }
#pragma unroll
  for (int nt = 0; nt < 4; nt++)
#pragma unroll
    for (int g = 0; g < 8; g++) {
      int trow = qt * 16 + g + rl;
      int u = h * CHD + nt * 16 + cl;
      obf[((size_t)b * CT + trow) * CU + u] = (__bf16)oa[nt][g];
    }
}

// ---------------- mean over T + concat with 'other' ----------------
__global__ __launch_bounds__(320)
void k_mean_concat(const float* __restrict__ xf, const float* __restrict__ other,
                   float* __restrict__ cf) {
  int b = blockIdx.x, u = threadIdx.x;
  if (u < CU) {
    float s = 0.f;
    for (int t = 0; t < CT; t++) s += xf[((size_t)b * CT + t) * CU + u];
    cf[b * 320 + u] = s * (1.f / CT);
  } else {
    cf[b * 320 + u] = other[b * 64 + (u - CU)];
  }
}

// ---------------- MLP head ----------------
__global__ __launch_bounds__(128)
void k_head(const float* __restrict__ cf, const float* __restrict__ m1W,
            const float* __restrict__ m1b, const float* __restrict__ l1s,
            const float* __restrict__ l1b, const float* __restrict__ m2W,
            const float* __restrict__ m2b, const float* __restrict__ l2s,
            const float* __restrict__ l2b, const float* __restrict__ oW,
            const float* __restrict__ ob, float* __restrict__ out) {
  __shared__ float c[320], h1[128], h2[64], red[32];
  int b = blockIdx.x, tid = threadIdx.x;
  for (int j = tid; j < 320; j += 128) c[j] = cf[b * 320 + j];
  __syncthreads();
  float acc = m1b[tid];
  for (int k = 0; k < 320; k++) acc += c[k] * m1W[k * 128 + tid];
  float v = fmaxf(acc, 0.f);
  float mean = block_sum(v, red) * (1.f / 128.f);
  float d = v - mean;
  float var = block_sum(d * d, red) * (1.f / 128.f);
  h1[tid] = d * rsqrtf(var + 1e-6f) * l1s[tid] + l1b[tid];
  __syncthreads();
  float v2 = 0.f, d2 = 0.f;
  if (tid < 64) {
    float a2 = m2b[tid];
    for (int k = 0; k < 128; k++) a2 += h1[k] * m2W[k * 64 + tid];
    v2 = fmaxf(a2, 0.f);
  }
  float mean2 = block_sum(v2, red) * (1.f / 64.f);
  if (tid < 64) d2 = v2 - mean2;
  float var2 = block_sum(d2 * d2, red) * (1.f / 64.f);
  if (tid < 64) h2[tid] = d2 * rsqrtf(var2 + 1e-6f) * l2s[tid] + l2b[tid];
  __syncthreads();
  float p = (tid < 64) ? h2[tid] * oW[tid] : 0.f;
  float s = block_sum(p, red);
  if (tid == 0) out[b] = s + ob[0];
}

// ---------------- launch ----------------
extern "C" void kernel_launch(void* const* d_in, const int* in_sizes, int n_in,
                              void* d_out, int out_size, void* d_ws, size_t ws_size,
                              hipStream_t stream) {
  const float* cgm  = (const float*)d_in[0];
  const float* other= (const float*)d_in[1];
  const float* d0W  = (const float*)d_in[2];
  const float* d0b  = (const float*)d_in[3];
  const float* ln0s = (const float*)d_in[4];
  const float* ln0b = (const float*)d_in[5];
  const float* gWi  = (const float*)d_in[6];
  const float* gbi  = (const float*)d_in[7];
  const float* gWh  = (const float*)d_in[8];
  const float* gbhn = (const float*)d_in[9];
  const float* ln1s = (const float*)d_in[10];
  const float* ln1b = (const float*)d_in[11];
  const float* Wq = (const float*)d_in[12]; const float* bq = (const float*)d_in[13];
  const float* Wk = (const float*)d_in[14]; const float* bk = (const float*)d_in[15];
  const float* Wv = (const float*)d_in[16]; const float* bv = (const float*)d_in[17];
  const float* Wo = (const float*)d_in[18]; const float* bo = (const float*)d_in[19];
  const float* ln2s = (const float*)d_in[20]; const float* ln2b = (const float*)d_in[21];
  const float* m1W = (const float*)d_in[22]; const float* m1b = (const float*)d_in[23];
  const float* l1s = (const float*)d_in[24]; const float* l1b = (const float*)d_in[25];
  const float* m2W = (const float*)d_in[26]; const float* m2b = (const float*)d_in[27];
  const float* l2s = (const float*)d_in[28]; const float* l2b = (const float*)d_in[29];
  const float* oW  = (const float*)d_in[30]; const float* ob  = (const float*)d_in[31];
  float* out = (float*)d_out;

  // workspace carve-out
  char* base = (char*)d_ws;
  size_t off = 0;
  auto take = [&](size_t bytes) -> char* {
    char* p = base + off;
    off += (bytes + 255) & ~(size_t)255;
    return p;
  };
  __bf16* WiT[2];   for (int i = 0; i < 2; i++) WiT[i]   = (__bf16*)take((size_t)C3U * CU * 2);
  __bf16* WhT[2];   for (int i = 0; i < 2; i++) WhT[i]   = (__bf16*)take((size_t)C3U * CU * 2);
  __bf16* WqkvT[2]; for (int i = 0; i < 2; i++) WqkvT[i] = (__bf16*)take((size_t)C3U * CU * 2);
  __bf16* WoT[2];   for (int i = 0; i < 2; i++) WoT[i]   = (__bf16*)take((size_t)CU * CU * 2);
  float*  bqkv = (float*)take(2 * C3U * sizeof(float));
  float*  xf   = (float*)take((size_t)CMT * CU * 4);
  __bf16* xbf  = (__bf16*)take((size_t)CMT * CU * 2);
  float*  xg   = (float*)take((size_t)CMT * C3U * 4);
  float*  gbuf = (float*)take((size_t)CMT * CU * 4);
  __bf16* qbf  = (__bf16*)take((size_t)CB * CH * CT * CHD * 2);
  __bf16* kbf  = (__bf16*)take((size_t)CB * CH * CT * CHD * 2);
  __bf16* vtbf = (__bf16*)take((size_t)CB * CH * CHD * CT * 2);
  __bf16* obf  = (__bf16*)take((size_t)CMT * CU * 2);
  float*  cf   = (float*)take((size_t)CB * 320 * 4);
  (void)ws_size; (void)in_sizes; (void)n_in; (void)out_size;

  // weight prep
  const int g3 = (C3U * CU + 255) / 256, g1 = (CU * CU + 255) / 256;
  for (int i = 0; i < 2; i++) {
    k_transpose_bf16<<<g3, 256, 0, stream>>>(gWi + (size_t)i * C3U * CU, WiT[i], CU, C3U);
    k_transpose_bf16<<<g3, 256, 0, stream>>>(gWh + (size_t)i * C3U * CU, WhT[i], CU, C3U);
    k_transpose_bf16<<<g1, 256, 0, stream>>>(Wq + (size_t)i * CU * CU, WqkvT[i], CU, CU);
    k_transpose_bf16<<<g1, 256, 0, stream>>>(Wk + (size_t)i * CU * CU, WqkvT[i] + (size_t)CU * CU, CU, CU);
    k_transpose_bf16<<<g1, 256, 0, stream>>>(Wv + (size_t)i * CU * CU, WqkvT[i] + (size_t)2 * CU * CU, CU, CU);
    k_transpose_bf16<<<g1, 256, 0, stream>>>(Wo + (size_t)i * CU * CU, WoT[i], CU, CU);
  }
  k_pack_qkv_bias<<<dim3(3, 2), 256, 0, stream>>>(bq, bk, bv, bqkv);

  // x = LN0(cgm @ d0_W + b)
  k_inproj_ln<<<CMT, 256, 0, stream>>>(cgm, d0W, d0b, ln0s, ln0b, xf, xbf);

  for (int i = 0; i < 2; i++) {
    // xg = x @ Wi + bi
    k_gemm_bf16<<<dim3(C3U / 64, CMT / 128), 256, 0, stream>>>(
        xbf, WiT[i], gbi + i * C3U, xg, nullptr, nullptr, nullptr, CMT, C3U, CU, 0);
    // GRU scan -> gbuf
    k_gru_scan<<<CB / 16, 512, 0, stream>>>(xg, WhT[i], gbhn + i * CU, gbuf);
    // x = LN1(g + x)
    k_add_ln<<<CMT, 256, 0, stream>>>(gbuf, xf, ln1s + i * CU, ln1b + i * CU, xf, xbf);
    // QKV (scattered bf16)
    k_gemm_bf16<<<dim3(C3U / 64, CMT / 128), 256, 0, stream>>>(
        xbf, WqkvT[i], bqkv + i * C3U, nullptr, qbf, kbf, vtbf, CMT, C3U, CU, 1);
    // attention -> obf
    k_attn<<<dim3(CB * CH, 3), 192, 0, stream>>>(qbf, kbf, vtbf, obf);
    // a = o @ Wo + bo -> gbuf
    k_gemm_bf16<<<dim3(CU / 64, CMT / 128), 256, 0, stream>>>(
        obf, WoT[i], bo + i * CU, gbuf, nullptr, nullptr, nullptr, CMT, CU, CU, 0);
    // x = LN2(a + x)
    k_add_ln<<<CMT, 256, 0, stream>>>(gbuf, xf, ln2s + i * CU, ln2b + i * CU, xf, xbf);
  }

  // mean over T, concat, head
  k_mean_concat<<<CB, 320, 0, stream>>>(xf, other, cf);
  k_head<<<CB, 128, 0, stream>>>(cf, m1W, m1b, l1s, l1b, m2W, m2b, l2s, l2b, oW, ob, out);
}
